// PCAA_9363028706004
// MI455X (gfx1250) — compile-verified, matrix operand order checked by reading, not run on hardware
//
#include <hip/hip_runtime.h>

// ---------------- types ----------------
typedef _Float16 h16;
typedef __attribute__((ext_vector_type(16))) _Float16 v16h;
typedef __attribute__((ext_vector_type(8)))  float    v8f;

#define BM 128
#define BN 128
#define BK 64
#define LDSS (BK + 8)   // padded stride (72): conflict-free b128 fragment gathers

// ---------------- epilogue params ----------------
struct EpiParams {
  const float* bias;        // per-col
  const float* row_scale;   // indexed [z*M + row] (bin_conf fold)
  float*       out_f32;
  size_t       out_f32_batch;
  h16*         out_h16;
  size_t       out_h16_batch;
  const float* bn_scale;
  const float* bn_shift;
  const float* resid;
  float*       final_out;
};

static inline EpiParams epi0() {
  EpiParams e;
  e.bias = nullptr; e.row_scale = nullptr;
  e.out_f32 = nullptr; e.out_f32_batch = 0;
  e.out_h16 = nullptr; e.out_h16_batch = 0;
  e.bn_scale = nullptr; e.bn_shift = nullptr;
  e.resid = nullptr; e.final_out = nullptr;
  return e;
}

// ---------------- WMMA GEMM: D[M,N] = A[M,K] * Bt[N,K]^T (both K-contiguous) ----------------
// MODE: 0 normal store, 1 patch-recover h16 store, 2 final BN+ReLU+residual
template<int MODE, bool RS, bool BIASF, bool OF32, bool OH16>
__global__ __launch_bounds__(256) void gemm_h16(
    const h16* __restrict__ A, size_t aBatch,
    const h16* __restrict__ B, size_t bBatch, int bDiv,
    int M, int N, int Kd, EpiParams ep)
{
  __shared__ __align__(16) h16 ldsA[BM * LDSS];
  __shared__ __align__(16) h16 ldsB[BN * LDSS];

  const int tid   = threadIdx.x;
  const int lane  = tid & 31;
  const int wid   = tid >> 5;
  const int waveM = wid & 3;       // 4 waves along M
  const int waveN = wid >> 2;      // 2 waves along N
  const int z     = blockIdx.z;
  const int tileN = blockIdx.x * BN;
  const int tileM = blockIdx.y * BM;
  const int l16   = lane & 15;
  const int hf    = lane >> 4;

  const h16* Ab = A + (size_t)z * aBatch + (size_t)tileM * Kd;
  const h16* Bb = B + (size_t)(z / bDiv) * bBatch + (size_t)tileN * Kd;

  v8f acc[2][4];
  #pragma unroll
  for (int mi = 0; mi < 2; ++mi)
    #pragma unroll
    for (int ni = 0; ni < 4; ++ni)
      #pragma unroll
      for (int j = 0; j < 8; ++j) acc[mi][ni][j] = 0.0f;

  for (int kt = 0; kt < Kd; kt += BK) {
    // global -> LDS: 1024 uint4 units per tile (A and B), 4 each per thread
    #pragma unroll
    for (int s = 0; s < 4; ++s) {
      int u   = tid + s * 256;
      int row = u >> 3;
      int c8  = u & 7;
      const uint4* ga = reinterpret_cast<const uint4*>(Ab + (size_t)row * Kd + kt) + c8;
      *reinterpret_cast<uint4*>(&ldsA[row * LDSS + c8 * 8]) = *ga;
      const uint4* gb = reinterpret_cast<const uint4*>(Bb + (size_t)row * Kd + kt) + c8;
      *reinterpret_cast<uint4*>(&ldsB[row * LDSS + c8 * 8]) = *gb;
    }
    __syncthreads();

    // 16-bit A fragment layout (16x32): lane half hf, row l16,
    // elems 0..7 -> k = hf*8+e ; elems 8..15 -> k = 16 + hf*8 + (e-8)
    #pragma unroll
    for (int kk = 0; kk < BK; kk += 32) {
      v16h afrag[2], bfrag[4];
      #pragma unroll
      for (int mi = 0; mi < 2; ++mi) {
        const h16* p = &ldsA[(waveM * 32 + mi * 16 + l16) * LDSS + kk + hf * 8];
        union { v16h v; h16 h[16]; } f;
        #pragma unroll
        for (int e = 0; e < 8; ++e) f.h[e] = p[e];
        #pragma unroll
        for (int e = 0; e < 8; ++e) f.h[8 + e] = p[16 + e];
        afrag[mi] = f.v;
      }
      #pragma unroll
      for (int ni = 0; ni < 4; ++ni) {
        const h16* p = &ldsB[(waveN * 64 + ni * 16 + l16) * LDSS + kk + hf * 8];
        union { v16h v; h16 h[16]; } f;
        #pragma unroll
        for (int e = 0; e < 8; ++e) f.h[e] = p[e];
        #pragma unroll
        for (int e = 0; e < 8; ++e) f.h[8 + e] = p[16 + e];
        bfrag[ni] = f.v;
      }
      #pragma unroll
      for (int mi = 0; mi < 2; ++mi)
        #pragma unroll
        for (int ni = 0; ni < 4; ++ni)
          acc[mi][ni] = __builtin_amdgcn_wmma_f32_16x16x32_f16(
              false, afrag[mi], false, bfrag[ni], (short)0, acc[mi][ni], false, false);
    }
    __syncthreads();
  }

  // ---- epilogue: C/D layout -> VGPR j holds (M = fragM0 + 8*hf + j, N = fragN0 + l16)
  #pragma unroll
  for (int mi = 0; mi < 2; ++mi) {
    #pragma unroll
    for (int ni = 0; ni < 4; ++ni) {
      const int col   = tileN + waveN * 64 + ni * 16 + l16;
      const int rbase = tileM + waveM * 32 + mi * 16 + hf * 8;
      float bcol = 0.0f;
      if constexpr (BIASF) bcol = ep.bias[col];
      float sc = 0.0f, sh = 0.0f;
      if constexpr (MODE == 2) { sc = ep.bn_scale[col]; sh = ep.bn_shift[col]; }
      #pragma unroll
      for (int j = 0; j < 8; ++j) {
        const int row = rbase + j;
        float v = acc[mi][ni][j];
        if constexpr (RS)    v *= ep.row_scale[(size_t)z * M + row];
        if constexpr (BIASF) v += bcol;
        if constexpr (MODE == 0) {
          const size_t idx = (size_t)row * N + col;
          if constexpr (OF32) ep.out_f32[(size_t)z * ep.out_f32_batch + idx] = v;
          if constexpr (OH16) ep.out_h16[(size_t)z * ep.out_h16_batch + idx] = (h16)v;
        } else if constexpr (MODE == 1) {
          // patch-recover store: z=(b,n), row=p -> [B,H,W,Ci] channel-last
          const int b = z >> 4, n = z & 15;
          const int h = ((n >> 2) << 5) + (row >> 5);
          const int w = ((n & 3) << 5) + (row & 31);
          const size_t dst = (((size_t)b * 16384) + (size_t)h * 128 + w) * (size_t)N + col;
          ep.out_h16[dst] = (h16)v;
        } else {
          // final: BN(eval) + ReLU + residual, channel-first scatter
          const int b = row >> 14;
          const int s = row & 16383;
          const size_t dst = ((size_t)b * 512 + col) * 16384 + s;
          ep.final_out[dst] = fmaxf(v * sc + sh, 0.0f) + ep.resid[dst];
        }
      }
    }
  }
}

// ---------------- aux kernels ----------------
// x [B,C,H,W] f32 -> x_p [B,bins,P,C] h16 and x_pT [B,bins,C,P] h16
__global__ void k_prep_xp(const float* __restrict__ x, h16* __restrict__ xp,
                          h16* __restrict__ xpT) {
  size_t i = (size_t)blockIdx.x * 256 + threadIdx.x;   // exactly B*C*H*W threads
  int w = (int)(i & 127);
  int h = (int)((i >> 7) & 127);
  int c = (int)((i >> 14) & 511);
  int b = (int)(i >> 23);
  int n = ((h >> 5) << 2) + (w >> 5);
  int p = ((h & 31) << 5) + (w & 31);
  h16 v = (h16)x[i];
  size_t bn = (size_t)(b * 16 + n);
  xp [(bn * 1024 + p) * 512 + c] = v;
  xpT[(bn * 512  + c) * 1024 + p] = v;
}

__global__ void k_cvt_f16(const float* __restrict__ s, h16* __restrict__ d, int n) {
  int i = blockIdx.x * 256 + threadIdx.x;
  if (i < n) d[i] = (h16)s[i];
}

__global__ void k_bnprep(const float* g, const float* beta, const float* mean,
                         const float* var, float* scale, float* shift) {
  int c = threadIdx.x;   // 512
  float s = g[c] * rsqrtf(var[c] + 1e-5f);
  scale[c] = s;
  shift[c] = beta[c] - mean[c] * s;
}

// fused: bin_conf[b,n,k] = sigmoid(mean_p cam_p[b,n,p,k]); conf_T[b,n,k,p] = softmax_p(cam_p)
__global__ void k_binstats(const float* __restrict__ cam, float* __restrict__ bconf,
                           h16* __restrict__ confT) {
  int bn = blockIdx.x;          // 64
  int k  = threadIdx.x;         // 512
  const float* base = cam + (size_t)bn * 1024 * 512 + k;
  float mx = -3.4e38f, sum = 0.f;
  for (int p = 0; p < 1024; ++p) {
    float t = base[(size_t)p * 512];
    mx = fmaxf(mx, t);
    sum += t;
  }
  bconf[bn * 512 + k] = 1.0f / (1.0f + expf(-sum * (1.0f / 1024.0f)));
  float se = 0.f;
  for (int p = 0; p < 1024; ++p) se += expf(base[(size_t)p * 512] - mx);
  float inv = 1.0f / se;
  h16* o = confT + ((size_t)bn * 512 + k) * 1024;
  for (int p = 0; p < 1024; ++p) o[p] = (h16)(expf(base[(size_t)p * 512] - mx) * inv);
}

// GCN over bins: g[b,n,k,c] = relu(sum_m w1[n,m]*local[b,m,k,c] + local[b,n,k,c]) -> h16
__global__ void k_gcn(const float* __restrict__ loc, const float* __restrict__ w1g,
                      h16* __restrict__ gout) {
  __shared__ float w1[256];
  int t = threadIdx.x;
  if (t < 256) w1[t] = w1g[t];
  __syncthreads();
  size_t i = (size_t)blockIdx.x * 256 + t;   // over (b,k,c)
  int c = (int)(i & 511), k = (int)((i >> 9) & 511), b = (int)(i >> 18);
  size_t base = (size_t)b * 16 * 262144 + (size_t)k * 512 + c;
  float v[16];
  #pragma unroll
  for (int m = 0; m < 16; ++m) v[m] = loc[base + (size_t)m * 262144];
  #pragma unroll
  for (int n = 0; n < 16; ++n) {
    float s = v[n];
    #pragma unroll
    for (int m = 0; m < 16; ++m) s += w1[n * 16 + m] * v[m];
    gout[base + (size_t)n * 262144] = (h16)fmaxf(s, 0.0f);
  }
}

// fuse: glob[b,k,c] = relu(b_fuse + sum_m w_fuse[m]*lf[b,m,k,c])
__global__ void k_fuse(const float* __restrict__ lf, const float* __restrict__ wf,
                       const float* __restrict__ bf, float* __restrict__ glob) {
  size_t i = (size_t)blockIdx.x * 256 + threadIdx.x;
  int c = (int)(i & 511), k = (int)((i >> 9) & 511), b = (int)(i >> 18);
  size_t base = (size_t)b * 16 * 262144 + (size_t)k * 512 + c;
  float s = bf[0];
  #pragma unroll
  for (int m = 0; m < 16; ++m) s += wf[m] * lf[base + (size_t)m * 262144];
  glob[((size_t)b * 512 + k) * 512 + c] = fmaxf(s, 0.0f);
}

// value (small GEMM, VALU): valueT[b,i,k] = b_v[i] + sum_c glob[b,k,c]*w_v[i,c]
__global__ void k_value(const float* __restrict__ glob, const float* __restrict__ wv,
                        const float* __restrict__ bv, h16* __restrict__ valT) {
  size_t idx = (size_t)blockIdx.x * 256 + threadIdx.x;  // (b,k,i)
  int ii = (int)(idx & 255), k = (int)((idx >> 8) & 511), b = (int)(idx >> 17);
  const float* g = glob + ((size_t)b * 512 + k) * 512;
  const float* w = wv + (size_t)ii * 512;
  float s = bv[ii];
  for (int c = 0; c < 512; ++c) s += g[c] * w[c];
  valT[((size_t)b * 256 + ii) * 512 + k] = (h16)s;
}

// row softmax over last dim (512), f32 in -> h16 out
__global__ void k_rowsoftmax(const float* __restrict__ in, h16* __restrict__ out) {
  __shared__ float red[256];
  int row = blockIdx.x, t = threadIdx.x;
  const float* r = in + (size_t)row * 512;
  float a = r[t], b = r[t + 256];
  red[t] = fmaxf(a, b);
  __syncthreads();
  for (int s = 128; s > 0; s >>= 1) {
    if (t < s) red[t] = fmaxf(red[t], red[t + s]);
    __syncthreads();
  }
  float mx = red[0];
  __syncthreads();
  float e0 = expf(a - mx), e1 = expf(b - mx);
  red[t] = e0 + e1;
  __syncthreads();
  for (int s = 128; s > 0; s >>= 1) {
    if (t < s) red[t] += red[t + s];
    __syncthreads();
  }
  float inv = 1.0f / red[0];
  out[(size_t)row * 512 + t]       = (h16)(e0 * inv);
  out[(size_t)row * 512 + t + 256] = (h16)(e1 * inv);
}

// ---------------- host launcher ----------------
extern "C" void kernel_launch(void* const* d_in, const int* in_sizes, int n_in,
                              void* d_out, int out_size, void* d_ws, size_t ws_size,
                              hipStream_t stream) {
  (void)in_sizes; (void)n_in; (void)out_size; (void)ws_size;
  const float* x      = (const float*)d_in[0];
  const float* w_cam  = (const float*)d_in[1];
  const float* b_cam  = (const float*)d_in[2];
  const float* w_gcn1 = (const float*)d_in[3];
  const float* w_gcn2 = (const float*)d_in[4];
  const float* w_fuse = (const float*)d_in[5];
  const float* b_fuse = (const float*)d_in[6];
  const float* w_q    = (const float*)d_in[7];
  const float* b_q    = (const float*)d_in[8];
  const float* w_k    = (const float*)d_in[9];
  const float* b_k    = (const float*)d_in[10];
  const float* w_v    = (const float*)d_in[11];
  const float* b_v    = (const float*)d_in[12];
  const float* w_out  = (const float*)d_in[13];
  const float* bn_g   = (const float*)d_in[14];
  const float* bn_b   = (const float*)d_in[15];
  const float* bn_m   = (const float*)d_in[16];
  const float* bn_v   = (const float*)d_in[17];
  float* out = (float*)d_out;
  char*  ws  = (char*)d_ws;

  // workspace layout (bytes), with buffer reuse
  h16*   xp    = (h16*)  (ws + (size_t)0);            // 64MB  x_p [B,bins,P,C]
  h16*   xpT   = (h16*)  (ws + (size_t)67108864);     // 64MB  x_pT (reused: aff_h16)
  float* cam   = (float*)(ws + (size_t)134217728);    // 128MB cam_p (reused: aff f32)
  h16*   confT = (h16*)  (ws + (size_t)268435456);    // 64MB  conf_T (reused: query/key/valueT)
  float* local = (float*)(ws + (size_t)335544320);    // 64MB  local (reused: local_feats f32)
  h16*   g     = (h16*)  (ws + (size_t)402653184);    // 32MB  g
  h16*   lfh   = (h16*)  (ws + (size_t)436207616);    // 32MB  local_feats h16
  h16*   outcl = (h16*)  (ws + (size_t)469762048);    // 32MB  out channel-last h16
  float* glob  = (float*)(ws + (size_t)503316480);    // 4MB
  float* bconf = (float*)(ws + (size_t)507510784);    // 128KB bin_conf
  h16*   wcamh = (h16*)  (ws + (size_t)507641856);
  h16*   wg2h  = (h16*)  (ws + (size_t)508166144);
  h16*   wqh   = (h16*)  (ws + (size_t)508690432);
  h16*   wkh   = (h16*)  (ws + (size_t)508952576);
  h16*   wouth = (h16*)  (ws + (size_t)509214720);
  float* bnsc  = (float*)(ws + (size_t)509476864);
  float* bnsh  = (float*)(ws + (size_t)509478912);
  float* lf    = local;                               // reuse (local dead after k_gcn)
  float* aff   = cam;                                 // reuse (cam dead after k_binstats)
  h16*   affh  = xpT;                                 // reuse (xpT dead after local GEMM)
  h16*   qh    = confT;                               // reuse (confT dead after local GEMM)
  h16*   kh    = (h16*)(ws + (size_t)268435456 + 33554432);
  h16*   valT  = (h16*)(ws + (size_t)268435456 + 50331648);

  dim3 blk(256);

  // prep: permute/convert x, weights, BN folding
  k_prep_xp<<<dim3(131072), blk, 0, stream>>>(x, xp, xpT);
  k_cvt_f16<<<dim3(1024), blk, 0, stream>>>(w_cam,  wcamh, 262144);
  k_cvt_f16<<<dim3(1024), blk, 0, stream>>>(w_gcn2, wg2h,  262144);
  k_cvt_f16<<<dim3(512),  blk, 0, stream>>>(w_q,    wqh,   131072);
  k_cvt_f16<<<dim3(512),  blk, 0, stream>>>(w_k,    wkh,   131072);
  k_cvt_f16<<<dim3(512),  blk, 0, stream>>>(w_out,  wouth, 131072);
  k_bnprep<<<dim3(1), dim3(512), 0, stream>>>(bn_g, bn_b, bn_m, bn_v, bnsc, bnsh);

  // 1) cam_p = x_p @ w_cam^T + b_cam   [65536,512] x [512,512]
  { EpiParams e = epi0(); e.bias = b_cam; e.out_f32 = cam;
    gemm_h16<0, false, true, true, false><<<dim3(4, 512, 1), blk, 0, stream>>>(
        xp, 0, wcamh, 0, 1, 65536, 512, 512, e); }

  // 2+3) bin_conf = sigmoid(avgpool), conf_T = softmax_p(cam_p) (fused, one pass fewer)
  k_binstats<<<dim3(64), dim3(512), 0, stream>>>(cam, bconf, confT);

  // 4) local = (conf^T @ x_p) * bin_conf   batched 64x [512,1024]x[1024,512]
  { EpiParams e = epi0(); e.row_scale = bconf; e.out_f32 = local; e.out_f32_batch = 262144;
    gemm_h16<0, true, false, true, false><<<dim3(4, 4, 64), blk, 0, stream>>>(
        confT, 524288, xpT, 524288, 1, 512, 512, 1024, e); }

  // 5) GCN mix over bins + relu -> g (h16)
  k_gcn<<<dim3(4096), blk, 0, stream>>>(local, w_gcn1, g);

  // 6) local_feats = g @ w_gcn2^T   [32768,512]x[512,512] -> f32 + h16
  { EpiParams e = epi0(); e.out_f32 = lf; e.out_h16 = lfh;
    gemm_h16<0, false, false, true, true><<<dim3(4, 256, 1), blk, 0, stream>>>(
        g, 0, wg2h, 0, 1, 32768, 512, 512, e); }

  // 7) fuse bins -> glob (relu)
  k_fuse<<<dim3(4096), blk, 0, stream>>>(lf, w_fuse, b_fuse, glob);

  // 8) query = x_p @ w_q^T + b_q   [65536,512]x[512,256] -> h16
  { EpiParams e = epi0(); e.bias = b_q; e.out_h16 = qh;
    gemm_h16<0, false, true, false, true><<<dim3(2, 512, 1), blk, 0, stream>>>(
        xp, 0, wqh, 0, 1, 65536, 256, 512, e); }

  // 9) key = local_feats @ w_k^T + b_k   [32768,512]x[512,256] -> h16
  { EpiParams e = epi0(); e.bias = b_k; e.out_h16 = kh;
    gemm_h16<0, false, true, false, true><<<dim3(2, 256, 1), blk, 0, stream>>>(
        lfh, 0, wkh, 0, 1, 32768, 256, 512, e); }

  // 10) valueT[b,i,k] = glob @ w_v^T + b_v (small, VALU)
  k_value<<<dim3(2048), blk, 0, stream>>>(glob, w_v, b_v, valT);

  // 11) aff = query @ key^T   batched 64x [1024,256]x[256,512] -> f32
  { EpiParams e = epi0(); e.out_f32 = aff; e.out_f32_batch = 524288;
    gemm_h16<0, false, false, true, false><<<dim3(4, 8, 64), blk, 0, stream>>>(
        qh, 262144, kh, 131072, 1, 1024, 512, 256, e); }

  // 12) softmax over classes -> h16
  k_rowsoftmax<<<dim3(65536), blk, 0, stream>>>(aff, affh);

  // 13) out = aff @ value   batched 64x [1024,512]x[512,256], patch-recover store
  { EpiParams e = epi0(); e.out_h16 = outcl;
    gemm_h16<1, false, false, false, true><<<dim3(2, 8, 64), blk, 0, stream>>>(
        affh, 524288, valT, 131072, 16, 1024, 256, 512, e); }

  // 14) final: y = out_cl @ w_out^T, BN + relu + residual, channel-first
  { EpiParams e = epi0(); e.bn_scale = bnsc; e.bn_shift = bnsh;
    e.resid = x; e.final_out = out;
    gemm_h16<2, false, false, false, false><<<dim3(4, 512, 1), blk, 0, stream>>>(
        outcl, 0, wouth, 0, 1, 65536, 512, 256, e); }
}
